// EdgeFeatureInjector_21045339750818
// MI455X (gfx1250) — matrix operand order repaired and checked by reading, not compiled, not run on hardware
//
#include <hip/hip_runtime.h>
#include <hip/hip_bf16.h>

typedef __attribute__((ext_vector_type(16))) _Float16 v16h;
typedef __attribute__((ext_vector_type(8)))  _Float16 v8h;
typedef __attribute__((ext_vector_type(8)))  float    v8f;
typedef __attribute__((ext_vector_type(4)))  float    v4f;

// ---------------------------------------------------------------------------
// out = x  (residual init; scatter-adds accumulate on top)
// ---------------------------------------------------------------------------
__global__ void init_out_kernel(const float* __restrict__ x,
                                float* __restrict__ out, int n4) {
    int i = blockIdx.x * blockDim.x + threadIdx.x;
    if (i < n4) ((float4*)out)[i] = ((const float4*)x)[i];
}

// ---------------------------------------------------------------------------
// Pre-pack W1 and W2 into exact per-lane WMMA B-fragment layout (f16).
//   w1p: [4 ntiles][32 lanes][16 halves]  (K=8 real rows, padded to 32 w/ 0)
//   w2p: [2 ksteps][4 ntiles][32 lanes][16 halves]
// B layout (32x16 f16): lane n<16 holds col N=n, K=kb+0..7 (halves 0..7),
// K=kb+16..23 (halves 8..15); lanes 16-31 hold col N=n-16, K=kb+8..15 / kb+24..31.
// ---------------------------------------------------------------------------
__global__ void pack_weights_kernel(const float* __restrict__ W1,   // [8,64]
                                    const float* __restrict__ W2,   // [64,64]
                                    _Float16* __restrict__ w1p,     // 2048 halves
                                    _Float16* __restrict__ w2p)     // 4096 halves
{
    int idx = blockIdx.x * blockDim.x + threadIdx.x;
    if (idx < 2048) {
        int nt   = idx >> 9;          // [0,4)
        int lane = (idx >> 4) & 31;
        int h    = idx & 15;
        int n    = nt * 16 + (lane & 15);
        float v  = (lane < 16 && h < 8) ? W1[h * 64 + n] : 0.0f;  // K>=8 -> 0
        w1p[idx] = (_Float16)v;
    }
    if (idx < 4096) {
        int ks   = idx >> 11;         // [0,2)
        int nt   = (idx >> 9) & 3;
        int lane = (idx >> 4) & 31;
        int h    = idx & 15;
        int K    = ks * 32 + ((h < 8) ? (8 * (lane >> 4) + h) : (8 + 8 * (lane >> 4) + h));
        int n    = nt * 16 + (lane & 15);
        w2p[idx] = (_Float16)W2[K * 64 + n];
    }
}

// ---------------------------------------------------------------------------
// Main kernel: 256 threads = 8 wave32; each wave owns 16 edges end-to-end.
// Both GEMMs on v_wmma_f32_16x16x32_f16; no workgroup barriers needed.
// ---------------------------------------------------------------------------
__global__ void __launch_bounds__(256)
edge_inject_kernel(const long long* __restrict__ edge_index, // [2, E] int64
                   const float*     __restrict__ edge_attr,  // [E, 8]
                   const _Float16*  __restrict__ w1p,        // packed B1 frags
                   const float*     __restrict__ b1,         // [64]
                   const _Float16*  __restrict__ w2p,        // packed B2 frags
                   const float*     __restrict__ b2,         // [64]
                   float*           __restrict__ out,        // [N,64] (= x)
                   const int E)
{
    constexpr int H_STRIDE = 72; // halves/row: 16B aligned, conflict-free b128
    __shared__ __align__(16) _Float16 h_lds[128 * H_STRIDE]; // 18 KB
    __shared__ float wind_lds[128];
    __shared__ int   dst_lds[128];

    const int tid  = threadIdx.x;
    const int lane = tid & 31;   // wave32
    const int wave = tid >> 5;
    const int ncol = lane & 15;
    const int hi   = lane >> 4;

    const long long eb = (long long)blockIdx.x * 128 + wave * 16;

    // ---- A1 fragment: edge_attr rows, K=8 padded to 32 ----
    // A layout (16x32 f16): lane<16 row M=lane, halves 0..7 = K0..7 (real),
    // halves 8..15 = K16..23 (pad 0); lanes>=16 = K8..15 / K24..31 (pad 0).
    v16h a1;
    #pragma unroll
    for (int i = 0; i < 16; ++i) a1[i] = (_Float16)0.0f;
    if (hi == 0) {
        const long long e  = eb + lane;
        const bool valid   = e < (long long)E;
        const long long es = valid ? e : 0;      // clamp tail; neutralized below
        const v4f* eap = (const v4f*)(edge_attr + es * 8);
        v4f lo4 = eap[0];
        v4f hi4 = eap[1];
        a1[0] = (_Float16)lo4[0]; a1[1] = (_Float16)lo4[1];
        a1[2] = (_Float16)lo4[2]; a1[3] = (_Float16)lo4[3];
        a1[4] = (_Float16)hi4[0]; a1[5] = (_Float16)hi4[1];
        a1[6] = (_Float16)hi4[2]; a1[7] = (_Float16)hi4[3];
        wind_lds[wave * 16 + lane] = valid ? (1.0f / (1.0f + __expf(-lo4[2]))) : 0.0f;
        dst_lds[wave * 16 + lane]  = valid ? (int)edge_index[(long long)E + e] : 0;
    }

    // ---- GEMM1: h = relu(ea @ W1 + b1) via 4 WMMAs ----
    v8f hacc[4];
    #pragma unroll
    for (int nt = 0; nt < 4; ++nt) {
        const float bv = b1[nt * 16 + ncol];
        #pragma unroll
        for (int i = 0; i < 8; ++i) hacc[nt][i] = bv;
    }
    #pragma unroll
    for (int nt = 0; nt < 4; ++nt) {
        v16h bf = *(const v16h*)(w1p + (nt * 32 + lane) * 16); // 32B contiguous
        hacc[nt] = __builtin_amdgcn_wmma_f32_16x16x32_f16(
            false, a1, false, bf, (short)0, hacc[nt], false, false);
    }

    // ---- relu + f32->f16, scatter C layout -> A layout through LDS ----
    // C/D: N = lane%16, M = vgpr + 8*hi  ->  h_lds[row][col]
    #pragma unroll
    for (int nt = 0; nt < 4; ++nt) {
        #pragma unroll
        for (int v = 0; v < 8; ++v) {
            float d = fmaxf(hacc[nt][v], 0.0f);
            h_lds[(wave * 16 + v + 8 * hi) * H_STRIDE + nt * 16 + ncol] = (_Float16)d;
        }
    }
    // Same-wave LDS RAW: LDS ops are in-order per wave; wait for stores.
    asm volatile("s_wait_dscnt 0x0" ::: "memory");

    // ---- A2 fragments from LDS (16x32 f16 layout, contiguous b128 loads) ----
    v16h afrag[2];
    {
        const int row = wave * 16 + ncol;
        #pragma unroll
        for (int ks = 0; ks < 2; ++ks) {
            const int off = row * H_STRIDE + ks * 32 + hi * 8;
            v8h l = *(const v8h*)&h_lds[off];
            v8h h = *(const v8h*)&h_lds[off + 16];
            v16h a;
            #pragma unroll
            for (int i = 0; i < 8; ++i) { a[i] = l[i]; a[i + 8] = h[i]; }
            afrag[ks] = a;
        }
    }

    // ---- GEMM2: edge_msg = (h @ W2 + b2) via 8 WMMAs ----
    v8f acc[4];
    #pragma unroll
    for (int nt = 0; nt < 4; ++nt) {
        const float bv = b2[nt * 16 + ncol];
        #pragma unroll
        for (int i = 0; i < 8; ++i) acc[nt][i] = bv;
    }
    #pragma unroll
    for (int ks = 0; ks < 2; ++ks) {
        #pragma unroll
        for (int nt = 0; nt < 4; ++nt) {
            v16h bf = *(const v16h*)(w2p + ((ks * 4 + nt) * 32 + lane) * 16);
            acc[nt] = __builtin_amdgcn_wmma_f32_16x16x32_f16(
                false, afrag[ks], false, bf, (short)0, acc[nt], false, false);
        }
    }

    // ---- Scatter: out[dst] += edge_msg * sigmoid(ea[2]) ----
    #pragma unroll
    for (int v = 0; v < 8; ++v) {
        const int m   = wave * 16 + v + 8 * hi;   // edge within block
        const float w = wind_lds[m];
        float* op = out + (long long)dst_lds[m] * 64;
        #pragma unroll
        for (int nt = 0; nt < 4; ++nt) {
            atomicAdd(op + nt * 16 + ncol, acc[nt][v] * w);
        }
    }
}

// ---------------------------------------------------------------------------
extern "C" void kernel_launch(void* const* d_in, const int* in_sizes, int n_in,
                              void* d_out, int out_size, void* d_ws, size_t ws_size,
                              hipStream_t stream) {
    const float*     x          = (const float*)d_in[0];
    const long long* edge_index = (const long long*)d_in[1];
    const float*     edge_attr  = (const float*)d_in[2];
    const float*     W1         = (const float*)d_in[3];
    const float*     b1         = (const float*)d_in[4];
    const float*     W2         = (const float*)d_in[5];
    const float*     b2         = (const float*)d_in[6];
    float*           out        = (float*)d_out;

    _Float16* w1p = (_Float16*)d_ws;        // 2048 halves
    _Float16* w2p = w1p + 2048;             // 4096 halves (12 KB total)

    const int E  = in_sizes[1] / 2;   // edge_index is [2, E]
    const int n4 = out_size / 4;      // N*64 floats -> float4 copies

    pack_weights_kernel<<<16, 256, 0, stream>>>(W1, W2, w1p, w2p);
    init_out_kernel<<<(n4 + 255) / 256, 256, 0, stream>>>(x, out, n4);

    const int nblocks = (E + 127) / 128;
    edge_inject_kernel<<<nblocks, 256, 0, stream>>>(
        edge_index, edge_attr, w1p, b1, w2p, b2, out, E);
}
